// DeformableConvBlock_53661321396395
// MI455X (gfx1250) — compile-verified, matrix operand order checked by reading, not run on hardware
//
#include <hip/hip_runtime.h>

#define BATCH 8
#define CIN   64
#define COUT  64
#define Hdim  128
#define Wdim  128
#define KK    576   // CIN*9
#define NKB   18    // KK/32
#define PIXT  32
#define PSTR  584   // padded LDS row stride (f16): 1168 B, 16B aligned, banks spread

typedef _Float16     v16h  __attribute__((ext_vector_type(16)));
typedef float        v8f   __attribute__((ext_vector_type(8)));
typedef float        f32x4 __attribute__((ext_vector_type(4)));
typedef unsigned int u32x4 __attribute__((ext_vector_type(4)));

union H8   { _Float16 h[8];  u32x4 q;    };
union FR   { u32x4 q[2];     v16h  v;    };   // A or B fragment (16 f16)
union F16U { f32x4 v[4];     float f[16];};

// ---------------- Kernel 0: NCHW -> NHWC tiled transpose ----------------
__global__ void nchw_to_nhwc(const float* __restrict__ x, float* __restrict__ xt) {
    __shared__ float tile[32][33];
    int tb  = blockIdx.x;            // B*H * 2 * 4 = 8192 blocks
    int bh  = tb >> 3;
    int sub = tb & 7;
    int c0  = (sub >> 2) * 32;
    int w0  = (sub & 3) * 32;
    int b   = bh >> 7, h = bh & 127;
    int lr  = threadIdx.x >> 5, lc = threadIdx.x & 31;
#pragma unroll
    for (int i = 0; i < 4; ++i) {
        int c = c0 + lr + i * 8;
        tile[lr + i * 8][lc] = x[(((b * CIN + c) * Hdim + h) * Wdim) + w0 + lc];
    }
    __syncthreads();
#pragma unroll
    for (int i = 0; i < 4; ++i) {
        int w = w0 + lr + i * 8;
        xt[(((b * Hdim + h) * Wdim + w) * CIN) + c0 + lc] = tile[lc][lr + i * 8];
    }
}

// ---- Kernel 0b: one-time weight permute+convert: [o][c*9+kt] f32 -> [o][kt*64+c] f16 ----
// blocks 0..63: main w -> wprep; blocks 64..95: w_off rows (>=18 zero) -> wprep_off
__global__ void prep_weights(const float* __restrict__ w, const float* __restrict__ w_off,
                             _Float16* __restrict__ wprep, _Float16* __restrict__ wprep_off) {
    int row = blockIdx.x;
    int c   = threadIdx.x;           // 0..63
    const float* src;
    _Float16* dst;
    bool zero = false;
    if (row < 64) { src = w + row * KK;            dst = wprep + row * KK; }
    else          { int r = row - 64;
                    src = w_off + r * KK;          dst = wprep_off + r * KK;
                    zero = (r >= 18); }
#pragma unroll
    for (int kt = 0; kt < 9; ++kt) {
        float v = zero ? 0.f : src[c * 9 + kt];
        dst[kt * 64 + c] = (_Float16)v;
    }
}

// ---------------- Kernel 1: fused offset-conv + deformable conv ----------------
__global__ void __launch_bounds__(128)
deform_fused(const float* __restrict__ xt,
             const _Float16* __restrict__ wprep, const _Float16* __restrict__ wprep_off,
             const float* __restrict__ b_off,    const float* __restrict__ bias,
             float* __restrict__ out) {
    __shared__ __align__(16) _Float16 P[PIXT][PSTR];   // patch / sampled matrix (f16)
    __shared__ float offs[PIXT][20];                   // 18 offset channels per pixel
    extern __shared__ __align__(16) _Float16 WS[];     // 64*PSTR: WOFF then WA (restaged)

    const int tid  = threadIdx.x;
    const int wave = tid >> 5;
    const int lane = tid & 31;
    const int n    = lane & 15;       // N index (pixel) / M row index for A
    const int kh   = lane >> 4;       // K-half select
    const int pixbase = blockIdx.x * PIXT;

    // (pixel, 16-channel group) mapping for data-movement phases
    const int pix = tid >> 2;
    const int cg  = tid & 3;
    const int p   = pixbase + pix;
    const int pb  = p >> 14;
    const int pho = (p >> 7) & 127;
    const int pwo = p & 127;

    // ---- Async stage W_off (pre-permuted f16, rows 18..31 pre-zeroed) -> WS rows 0..31 ----
    {
        const int row = tid >> 2;          // 0..31
        const int q   = tid & 3;           // 18 16B-chunks each (72 per 1152B row)
        const char* g = (const char*)(wprep_off + row * KK) + q * 18 * 16;
        unsigned    l = (unsigned)(size_t)(WS + row * PSTR) + q * 18 * 16;
        for (int i = 0; i < 18; ++i) {
            asm volatile("global_load_async_to_lds_b128 %0, %1, off"
                         :: "v"(l + i * 16), "v"(g + i * 16) : "memory");
        }
    }

    // ---- Phase A: regular 3x3 patch -> P (overlaps with async weight DMA) ----
    for (int k = 0; k < 9; ++k) {
        int ki = k / 3, kj = k % 3;
        int py = pho - 1 + ki, px = pwo - 1 + kj;
        bool valid = ((unsigned)py < (unsigned)Hdim) && ((unsigned)px < (unsigned)Wdim);
        F16U buf;
        if (valid) {
            const f32x4* src = (const f32x4*)(xt + (((pb * Hdim + py) * Wdim + px) * CIN) + cg * 16);
            buf.v[0] = src[0]; buf.v[1] = src[1]; buf.v[2] = src[2]; buf.v[3] = src[3];
        } else {
#pragma unroll
            for (int j = 0; j < 16; ++j) buf.f[j] = 0.f;
        }
        H8 lo, hi;
#pragma unroll
        for (int j = 0; j < 8; ++j) { lo.h[j] = (_Float16)buf.f[j]; hi.h[j] = (_Float16)buf.f[8 + j]; }
        u32x4* dst = (u32x4*)&P[pix][k * 64 + cg * 16];
        dst[0] = lo.q; dst[1] = hi.q;
    }
    asm volatile("s_wait_asynccnt 0x0" ::: "memory");
    __syncthreads();

    // ---- Phase B: offset conv GEMM (M=18 padded to 32), WMMA f16->f32 ----
    {
        const int mt = wave >> 1;     // M tile 0..1
        const int nt = wave & 1;      // N tile 0..1
        v8f acc = {};
        for (int kb = 0; kb < NKB; ++kb) {
            FR a, bf;
            const u32x4* as = (const u32x4*)(WS + (mt * 16 + n) * PSTR + kb * 32 + kh * 8);
            a.q[0] = as[0]; a.q[1] = as[2];          // K rel {0..7},{16..23} (+8 if kh)
            const u32x4* bs = (const u32x4*)&P[nt * 16 + n][kb * 32 + kh * 8];
            bf.q[0] = bs[0]; bf.q[1] = bs[2];
            acc = __builtin_amdgcn_wmma_f32_16x16x32_f16(false, a.v, false, bf.v,
                                                         (short)0, acc, false, false);
        }
#pragma unroll
        for (int j = 0; j < 8; ++j) {
            int r = mt * 16 + j + kh * 8;
            if (r < 18) offs[nt * 16 + n][r] = acc[j] + b_off[r];
        }
    }
    __syncthreads();

    // ---- Async stage main W (pre-permuted f16) -> WS rows 0..63 (overwrites WOFF) ----
    // DMA overlaps the whole bilinear sampling phase below.
    {
        const int row  = tid >> 1;         // 0..63
        const int half = tid & 1;          // 36 16B-chunks each
        const char* g = (const char*)(wprep + row * KK) + half * 36 * 16;
        unsigned    l = (unsigned)(size_t)(WS + row * PSTR) + half * 36 * 16;
        for (int i = 0; i < 36; ++i) {
            asm volatile("global_load_async_to_lds_b128 %0, %1, off"
                         :: "v"(l + i * 16), "v"(g + i * 16) : "memory");
        }
    }

    // ---- Phase C: bilinear deformable sampling -> P (overwrites patch) ----
    for (int k = 0; k < 9; ++k) {
        int ki = k / 3, kj = k % 3;
        float py = (float)(pho - 1 + ki) + offs[pix][2 * k];
        float px = (float)(pwo - 1 + kj) + offs[pix][2 * k + 1];
        float y0 = floorf(py), x0 = floorf(px);
        float wy = py - y0, wx = px - x0;
        int iy0 = (int)y0, ix0 = (int)x0;
        float accv[16];
#pragma unroll
        for (int j = 0; j < 16; ++j) accv[j] = 0.f;
#pragma unroll
        for (int dy = 0; dy < 2; ++dy)
#pragma unroll
            for (int dx = 0; dx < 2; ++dx) {
                int yy = iy0 + dy, xx = ix0 + dx;
                bool valid = ((unsigned)yy < (unsigned)Hdim) && ((unsigned)xx < (unsigned)Wdim);
                float wgt = (dy ? wy : 1.f - wy) * (dx ? wx : 1.f - wx);
                wgt = valid ? wgt : 0.f;
                int yc = yy < 0 ? 0 : (yy > Hdim - 1 ? Hdim - 1 : yy);
                int xc = xx < 0 ? 0 : (xx > Wdim - 1 ? Wdim - 1 : xx);
                const f32x4* g = (const f32x4*)(xt + (((pb * Hdim + yc) * Wdim + xc) * CIN) + cg * 16);
                F16U buf; buf.v[0] = g[0]; buf.v[1] = g[1]; buf.v[2] = g[2]; buf.v[3] = g[3];
#pragma unroll
                for (int j = 0; j < 16; ++j) accv[j] += buf.f[j] * wgt;
            }
        H8 lo, hi;
#pragma unroll
        for (int j = 0; j < 8; ++j) { lo.h[j] = (_Float16)accv[j]; hi.h[j] = (_Float16)accv[8 + j]; }
        u32x4* dst = (u32x4*)&P[pix][k * 64 + cg * 16];
        dst[0] = lo.q; dst[1] = hi.q;
    }
    asm volatile("s_wait_asynccnt 0x0" ::: "memory");
    __syncthreads();

    // ---- Phase D: main GEMM (64 x 576) x (576 x 32), WMMA f16->f32 ----
    v8f acc0 = {}, acc1 = {};
    const _Float16* warow = WS + (wave * 16 + n) * PSTR;   // Cout row for A
    for (int kb = 0; kb < NKB; ++kb) {
        FR a, b0, b1;
        const u32x4* as = (const u32x4*)(warow + kb * 32 + kh * 8);
        a.q[0] = as[0]; a.q[1] = as[2];
        const u32x4* s0 = (const u32x4*)&P[n][kb * 32 + kh * 8];
        const u32x4* s1 = (const u32x4*)&P[16 + n][kb * 32 + kh * 8];
        b0.q[0] = s0[0]; b0.q[1] = s0[2];
        b1.q[0] = s1[0]; b1.q[1] = s1[2];
        acc0 = __builtin_amdgcn_wmma_f32_16x16x32_f16(false, a.v, false, b0.v,
                                                      (short)0, acc0, false, false);
        acc1 = __builtin_amdgcn_wmma_f32_16x16x32_f16(false, a.v, false, b1.v,
                                                      (short)0, acc1, false, false);
    }

    // ---- Phase E: bias + store NCHW (coalesced along wo) ----
#pragma unroll
    for (int j = 0; j < 8; ++j) {
        int cout = wave * 16 + j + kh * 8;
        float bv = bias[cout];
#pragma unroll
        for (int nt = 0; nt < 2; ++nt) {
            int pp = pixbase + nt * 16 + n;
            int ob = pp >> 14, oh = (pp >> 7) & 127, ow = pp & 127;
            float r = (nt ? acc1[j] : acc0[j]) + bv;
            out[(((ob * COUT + cout) * Hdim + oh) * Wdim) + ow] = r;
        }
    }
}

extern "C" void kernel_launch(void* const* d_in, const int* in_sizes, int n_in,
                              void* d_out, int out_size, void* d_ws, size_t ws_size,
                              hipStream_t stream) {
    const float* x     = (const float*)d_in[0];
    const float* w_off = (const float*)d_in[1];
    const float* b_off = (const float*)d_in[2];
    const float* w     = (const float*)d_in[3];
    const float* b     = (const float*)d_in[4];
    float* out = (float*)d_out;

    // workspace: [0,32MB) xt NHWC f32 ; then wprep (64*576 f16) ; then wprep_off (32*576 f16)
    char* ws = (char*)d_ws;
    float*     xt        = (float*)ws;
    _Float16*  wprep     = (_Float16*)(ws + (size_t)BATCH * Hdim * Wdim * CIN * 4);
    _Float16*  wprep_off = wprep + (size_t)64 * KK;

    nchw_to_nhwc<<<8192, 256, 0, stream>>>(x, xt);
    prep_weights<<<96, 64, 0, stream>>>(w, w_off, wprep, wprep_off);
    const size_t ws_lds = (size_t)64 * PSTR * sizeof(_Float16);   // 74752 B dynamic LDS
    deform_fused<<<4096, 128, ws_lds, stream>>>(xt, wprep, wprep_off, b_off, b, out);
}
